// MoE_39917426049111
// MI455X (gfx1250) — compile-verified, matrix operand order checked by reading, not run on hardware
//
#include <hip/hip_runtime.h>
#include <cstdint>
#include <cstddef>

typedef __attribute__((ext_vector_type(8))) int v8i;

constexpr int E_   = 8;
constexpr int NTOK = 8192;   // B*S
constexpr int D_   = 1024;
constexpr int H_   = 2048;
constexpr int H2_  = 4096;   // 2H

constexpr int KC      = 256;          // K-chunk bytes staged in LDS
constexpr int LDS_PAD = 16;           // per-row pad: stride 272 -> 4-bank row skew
constexpr int BSTRIDE = KC + LDS_PAD; // 272
constexpr int KT      = KC / 64;      // 4 WMMA k-steps per chunk

__device__ __forceinline__ v8i zero_v8i() {
  v8i z;
#pragma unroll
  for (int i = 0; i < 8; ++i) z[i] = 0;
  return z;
}

// 32 contiguous bytes -> v8i (two b128 loads; works for global and LDS ptrs)
__device__ __forceinline__ v8i load_v8i(const signed char* p) {
  const int4* q = reinterpret_cast<const int4*>(p);
  int4 lo = q[0];
  int4 hi = q[1];
  v8i r;
  r[0] = lo.x; r[1] = lo.y; r[2] = lo.z; r[3] = lo.w;
  r[4] = hi.x; r[5] = hi.y; r[6] = hi.z; r[7] = hi.w;
  return r;
}

// ---------------------------------------------------------------------------
// Router: one wave32 per token. Logits (8 dots over D), top-2 + gates,
// loss accumulators, compacted per-expert token lists, int8 act-quant.
// ---------------------------------------------------------------------------
__global__ __launch_bounds__(256) void router_kernel(
    const float* __restrict__ x, const float* __restrict__ wgate,
    signed char* __restrict__ xq, float* __restrict__ step_x,
    int* __restrict__ counts, int* __restrict__ list_tok,
    float* __restrict__ list_gate, float* __restrict__ p_acc,
    float* __restrict__ freq_acc, float* __restrict__ lse2_acc) {
  const int lane = threadIdx.x & 31;
  const int wv   = threadIdx.x >> 5;
  const int t    = blockIdx.x * 8 + wv;
  const float* xrow = x + (size_t)t * D_;

  float lg[8];
#pragma unroll
  for (int i = 0; i < 8; ++i) lg[i] = 0.f;
  float amax = 0.f;

  for (int d = lane; d < D_; d += 32) {
    float xv = xrow[d];
    amax = fmaxf(amax, fabsf(xv));
    const float4* wg4 = reinterpret_cast<const float4*>(wgate + (size_t)d * 8);
    float4 wa = wg4[0], wb = wg4[1];
    lg[0] += xv * wa.x; lg[1] += xv * wa.y; lg[2] += xv * wa.z; lg[3] += xv * wa.w;
    lg[4] += xv * wb.x; lg[5] += xv * wb.y; lg[6] += xv * wb.z; lg[7] += xv * wb.w;
  }
#pragma unroll
  for (int m = 16; m >= 1; m >>= 1) {
    amax = fmaxf(amax, __shfl_xor(amax, m, 32));
#pragma unroll
    for (int i = 0; i < 8; ++i) lg[i] += __shfl_xor(lg[i], m, 32);
  }

  float cm = fmaxf(amax, 1e-5f);
  float scale = 127.f / cm;

  if (lane == 0) {
    step_x[t] = cm / 127.f;
    int i0 = 0; float v0 = lg[0];
#pragma unroll
    for (int i = 1; i < 8; ++i) if (lg[i] > v0) { v0 = lg[i]; i0 = i; }
    int i1 = (i0 == 0) ? 1 : 0; float v1 = lg[i1];
#pragma unroll
    for (int i = 0; i < 8; ++i) if (i != i0 && lg[i] > v1) { v1 = lg[i]; i1 = i; }
    float e1 = __expf(v1 - v0);
    float gs = 1.f + e1;
    float g0 = 1.f / gs, g1 = e1 / gs;
    float se = 0.f;
#pragma unroll
    for (int i = 0; i < 8; ++i) se += __expf(lg[i] - v0);
    float lse = __logf(se) + v0;
    atomicAdd(lse2_acc, lse * lse);
#pragma unroll
    for (int i = 0; i < 8; ++i) atomicAdd(&p_acc[i], __expf(lg[i] - v0) / se);
    atomicAdd(&freq_acc[i0], 1.f);
    atomicAdd(&freq_acc[i1], 1.f);
    int p0 = atomicAdd(&counts[i0], 1);
    list_tok[i0 * NTOK + p0]  = t;
    list_gate[i0 * NTOK + p0] = g0;
    int p1 = atomicAdd(&counts[i1], 1);
    list_tok[i1 * NTOK + p1]  = t;
    list_gate[i1 * NTOK + p1] = g1;
  }

  for (int d = lane; d < D_; d += 32) {
    float q = rintf(xrow[d] * scale);
    q = fminf(fmaxf(q, -128.f), 127.f);
    xq[(size_t)t * D_ + d] = (signed char)(int)q;
  }
}

// Per-expert |w| sum reduction. grid = (blocksPerExpert, E)
__global__ __launch_bounds__(256) void wabs_kernel(const float* __restrict__ w,
                                                   float* __restrict__ acc,
                                                   int perExpert) {
  const int e = blockIdx.y;
  const float* base = w + (size_t)e * perExpert;
  float s = 0.f;
  for (int i = blockIdx.x * 256 + threadIdx.x; i < perExpert; i += gridDim.x * 256)
    s += fabsf(base[i]);
#pragma unroll
  for (int m = 16; m >= 1; m >>= 1) s += __shfl_xor(s, m, 32);
  __shared__ float red[8];
  if ((threadIdx.x & 31) == 0) red[threadIdx.x >> 5] = s;
  __syncthreads();
  if (threadIdx.x == 0) {
    float tot = 0.f;
#pragma unroll
    for (int i = 0; i < 8; ++i) tot += red[i];
    atomicAdd(&acc[e], tot);
  }
}

// Ternarize weights to int8 in {-1,0,1}; real value = q * step_w[e].
__global__ __launch_bounds__(256) void wquant_kernel(const float* __restrict__ w,
                                                     signed char* __restrict__ wq,
                                                     const float* __restrict__ wsum,
                                                     int perExpert) {
  const int e = blockIdx.y;
  float mean  = wsum[e] / (float)perExpert;
  float scale = 1.f / fmaxf(mean, 1e-5f);
  const float* base = w + (size_t)e * perExpert;
  signed char* dst  = wq + (size_t)e * perExpert;
  for (int i = blockIdx.x * 256 + threadIdx.x; i < perExpert; i += gridDim.x * 256) {
    float q = rintf(base[i] * scale);
    q = fminf(fmaxf(q, -1.f), 1.f);
    dst[i] = (signed char)(int)q;
  }
}

// ---------------------------------------------------------------------------
// Grouped GEMM1 + GLU:  h = xq[list] @ Win[e]^T   (IU8 WMMA)
// Block = 8 waves = 4 row-groups x 2 col-groups over 128 routed tokens and a
// 64-col n_tile (+ paired +H gate group). Each wave: 32 rows x 32 cols(+gate),
// so every LDS B fragment feeds TWO independent WMMAs (halved ds traffic).
// B panel (128 rows) staged in LDS in 256B K-chunks; A gathered from global
// with next-k software prefetch.
// ---------------------------------------------------------------------------
__global__ __launch_bounds__(256) void gemm1_glu_kernel(
    const signed char* __restrict__ xq, const float* __restrict__ step_x,
    const signed char* __restrict__ winq, const float* __restrict__ wsum_in,
    const int* __restrict__ counts, const int* __restrict__ list_tok,
    float* __restrict__ hbuf, int e) {
  constexpr int NT = H_ / 64;  // 32 column tiles
  const int cnt    = counts[e];
  const int n_tile = blockIdx.x & (NT - 1);
  const int m_blk  = blockIdx.x >> 5;
  if (m_blk * 128 >= cnt) return;  // block-uniform exit

  __shared__ alignas(16) signed char ldsB[128 * BSTRIDE];  // ~34 KB

  const int lane  = threadIdx.x & 31;
  const int wv    = threadIdx.x >> 5;
  const int wv_m  = wv >> 1;   // row group 0..3 (32 rows each)
  const int wv_c  = wv & 1;    // col group 0..1 (32 cols each)
  const int l15   = lane & 15;
  const int khalf = lane >> 4;
  const int n0    = n_tile * 64;

  // Two gathered A rows per lane (two 16-row subtiles); clamped past cnt.
  const int rbase = m_blk * 128 + wv_m * 32;
  int r0 = rbase + l15;
  int r1 = rbase + 16 + l15;
  int rc0 = (r0 < cnt) ? r0 : cnt - 1;
  int rc1 = (r1 < cnt) ? r1 : cnt - 1;
  const signed char* aptr0 =
      xq + (size_t)list_tok[e * NTOK + rc0] * D_ + khalf * 32;
  const signed char* aptr1 =
      xq + (size_t)list_tok[e * NTOK + rc1] * D_ + khalf * 32;

  const signed char* wbase = winq + (size_t)e * H2_ * D_;

  // acc[mg*4 + g*2 + j2]: mg=row group, g=0 h1 / g=1 gate, j2=16-col subtile
  v8i acc[8];
#pragma unroll
  for (int j = 0; j < 8; ++j) acc[j] = zero_v8i();

  for (int kc = 0; kc < D_ / KC; ++kc) {
    if (kc) __syncthreads();
    // cooperative fill: 128 rows x 256 B (rows 0-63: h1 group, 64-127: gate)
#pragma unroll
    for (int i = threadIdx.x * 16; i < 128 * KC; i += 256 * 16) {
      int row  = i >> 8;       // /256
      int col  = i & (KC - 1);
      int grow = (row < 64) ? (n0 + row) : (H_ + n0 + row - 64);
      int4 v = *reinterpret_cast<const int4*>(wbase + (size_t)grow * D_ +
                                              kc * KC + col);
      *reinterpret_cast<int4*>(&ldsB[row * BSTRIDE + col]) = v;
    }
    __syncthreads();

    v8i a_cur0 = load_v8i(aptr0 + kc * KC);
    v8i a_cur1 = load_v8i(aptr1 + kc * KC);

#pragma unroll
    for (int kt = 0; kt < KT; ++kt) {
      // 4 B fragments for this wave's 32(+32 gate) cols
      v8i bf[4];
#pragma unroll
      for (int g = 0; g < 2; ++g)
#pragma unroll
        for (int j2 = 0; j2 < 2; ++j2) {
          int brow = g * 64 + wv_c * 32 + j2 * 16 + l15;
          bf[g * 2 + j2] =
              load_v8i(&ldsB[brow * BSTRIDE + khalf * 32 + kt * 64]);
        }
      // software-prefetch next k-step's A fragments (global)
      v8i a_nxt0 = zero_v8i(), a_nxt1 = zero_v8i();
      if (kt + 1 < KT) {
        a_nxt0 = load_v8i(aptr0 + kc * KC + (kt + 1) * 64);
        a_nxt1 = load_v8i(aptr1 + kc * KC + (kt + 1) * 64);
      }
      // each bf feeds two independent WMMAs (row groups 0 and 1)
#pragma unroll
      for (int q = 0; q < 4; ++q) {
        acc[q]     = __builtin_amdgcn_wmma_i32_16x16x64_iu8(
            true, a_cur0, true, bf[q], acc[q], false, false);
        acc[4 + q] = __builtin_amdgcn_wmma_i32_16x16x64_iu8(
            true, a_cur1, true, bf[q], acc[4 + q], false, false);
      }
      if (kt + 1 < KT) { a_cur0 = a_nxt0; a_cur1 = a_nxt1; }
    }
  }

  const float sw = fmaxf(wsum_in[e] / (float)(H2_ * D_), 1e-5f);
#pragma unroll
  for (int mg = 0; mg < 2; ++mg) {
#pragma unroll
    for (int v = 0; v < 8; ++v) {
      int m   = (khalf << 3) + v;  // C layout: lanes 0-15 rows 0-7, 16-31 rows 8-15
      int row = rbase + mg * 16 + m;
      if (row < cnt) {
        int tok = list_tok[e * NTOK + row];
        float f = step_x[tok] * sw;
#pragma unroll
        for (int j2 = 0; j2 < 2; ++j2) {
          float h1 = (float)acc[mg * 4 + j2][v] * f;
          float g  = (float)acc[mg * 4 + 2 + j2][v] * f;
          float si = h1 / (1.f + __expf(-h1));
          hbuf[(size_t)row * H_ + n0 + wv_c * 32 + j2 * 16 + l15] = si * g;
        }
      }
    }
  }
}

// Per-routed-row absmax requant of the GLU output to int8.
__global__ __launch_bounds__(256) void hquant_kernel(
    const float* __restrict__ hbuf, signed char* __restrict__ hq,
    float* __restrict__ step_h, const int* __restrict__ counts, int e) {
  const int cnt  = counts[e];
  const int lane = threadIdx.x & 31;
  const int wv   = threadIdx.x >> 5;
  const int row  = blockIdx.x * 8 + wv;
  if (row >= cnt) return;
  const float* src = hbuf + (size_t)row * H_;
  float mx = 0.f;
  for (int i = lane; i < H_; i += 32) mx = fmaxf(mx, fabsf(src[i]));
#pragma unroll
  for (int m = 16; m >= 1; m >>= 1) mx = fmaxf(mx, __shfl_xor(mx, m, 32));
  float cm = fmaxf(mx, 1e-5f);
  float scale = 127.f / cm;
  if (lane == 0) step_h[row] = cm / 127.f;
  signed char* dst = hq + (size_t)row * H_;
  for (int i = lane; i < H_; i += 32) {
    float q = rintf(src[i] * scale);
    q = fminf(fmaxf(q, -128.f), 127.f);
    dst[i] = (signed char)(int)q;
  }
}

// ---------------------------------------------------------------------------
// Grouped GEMM2: y[token] += gate * (hq @ Wout[e]^T) * step_h * step_w.
// Block = 8 waves x 32 rows = 256 routed rows sharing one 64-col B panel.
// Each wave: 32 rows x 64 cols -> every B fragment feeds two WMMAs.
// ---------------------------------------------------------------------------
__global__ __launch_bounds__(256) void gemm2_kernel(
    const signed char* __restrict__ hq, const float* __restrict__ step_h,
    const signed char* __restrict__ woutq, const float* __restrict__ wsum_out,
    const int* __restrict__ counts, const int* __restrict__ list_tok,
    const float* __restrict__ list_gate, float* __restrict__ y, int e) {
  constexpr int NT = D_ / 64;  // 16 column tiles
  const int cnt    = counts[e];
  const int n_tile = blockIdx.x & (NT - 1);
  const int m_blk  = blockIdx.x >> 4;
  if (m_blk * 256 >= cnt) return;

  __shared__ alignas(16) signed char ldsB[64 * BSTRIDE];  // ~17 KB

  const int lane  = threadIdx.x & 31;
  const int wv    = threadIdx.x >> 5;
  const int l15   = lane & 15;
  const int khalf = lane >> 4;
  const int n0    = n_tile * 64;

  const int rbase = m_blk * 256 + wv * 32;
  int r0 = rbase + l15;
  int r1 = rbase + 16 + l15;
  int rc0 = (r0 < cnt) ? r0 : cnt - 1;
  int rc1 = (r1 < cnt) ? r1 : cnt - 1;
  const signed char* aptr0 = hq + (size_t)rc0 * H_ + khalf * 32;
  const signed char* aptr1 = hq + (size_t)rc1 * H_ + khalf * 32;
  const signed char* wbase = woutq + (size_t)e * D_ * H_;

  // acc[mg*4 + j]: mg=row group, j=16-col subtile of the 64-col tile
  v8i acc[8];
#pragma unroll
  for (int j = 0; j < 8; ++j) acc[j] = zero_v8i();

  for (int kc = 0; kc < H_ / KC; ++kc) {
    if (kc) __syncthreads();
#pragma unroll
    for (int i = threadIdx.x * 16; i < 64 * KC; i += 256 * 16) {
      int row = i >> 8;
      int col = i & (KC - 1);
      int4 v = *reinterpret_cast<const int4*>(wbase + (size_t)(n0 + row) * H_ +
                                              kc * KC + col);
      *reinterpret_cast<int4*>(&ldsB[row * BSTRIDE + col]) = v;
    }
    __syncthreads();

    v8i a_cur0 = load_v8i(aptr0 + kc * KC);
    v8i a_cur1 = load_v8i(aptr1 + kc * KC);

#pragma unroll
    for (int kt = 0; kt < KT; ++kt) {
      v8i bf[4];
#pragma unroll
      for (int j = 0; j < 4; ++j) {
        int brow = j * 16 + l15;
        bf[j] = load_v8i(&ldsB[brow * BSTRIDE + khalf * 32 + kt * 64]);
      }
      v8i a_nxt0 = zero_v8i(), a_nxt1 = zero_v8i();
      if (kt + 1 < KT) {
        a_nxt0 = load_v8i(aptr0 + kc * KC + (kt + 1) * 64);
        a_nxt1 = load_v8i(aptr1 + kc * KC + (kt + 1) * 64);
      }
#pragma unroll
      for (int j = 0; j < 4; ++j) {
        acc[j]     = __builtin_amdgcn_wmma_i32_16x16x64_iu8(
            true, a_cur0, true, bf[j], acc[j], false, false);
        acc[4 + j] = __builtin_amdgcn_wmma_i32_16x16x64_iu8(
            true, a_cur1, true, bf[j], acc[4 + j], false, false);
      }
      if (kt + 1 < KT) { a_cur0 = a_nxt0; a_cur1 = a_nxt1; }
    }
  }

  const float sw = fmaxf(wsum_out[e] / (float)(D_ * H_), 1e-5f);
#pragma unroll
  for (int mg = 0; mg < 2; ++mg) {
#pragma unroll
    for (int v = 0; v < 8; ++v) {
      int m   = (khalf << 3) + v;
      int row = rbase + mg * 16 + m;
      if (row < cnt) {
        int tok = list_tok[e * NTOK + row];
        float f = step_h[row] * sw * list_gate[e * NTOK + row];
#pragma unroll
        for (int j = 0; j < 4; ++j) {
          size_t oi = (size_t)tok * D_ + n0 + j * 16 + l15;
          y[oi] += (float)acc[mg * 4 + j][v] * f;  // experts serialized: race-free
        }
      }
    }
  }
}

__global__ __launch_bounds__(256) void init_y_kernel(float* __restrict__ y,
                                                     const float* __restrict__ bias) {
  size_t i = (size_t)blockIdx.x * 256 + threadIdx.x;
  if (i < (size_t)NTOK * D_) y[i] = bias[i & (D_ - 1)];
}

__global__ void zero_accum_kernel(float* __restrict__ w) {
  if (threadIdx.x < 48) w[threadIdx.x] = 0.f;
}

__global__ void loss_kernel(const float* __restrict__ p_acc,
                            const float* __restrict__ freq_acc,
                            const float* __restrict__ lse2,
                            float* __restrict__ out) {
  float sp = 0.f, sf = 0.f;
  for (int e = 0; e < 8; ++e) { sp += p_acc[e]; sf += freq_acc[e]; }
  float sl = 0.f;
  for (int e = 0; e < 8; ++e) sl += (p_acc[e] / sp) * (freq_acc[e] / sf);
  out[(size_t)NTOK * D_] = 8.f * sl + 0.1f * (lse2[0] / (float)NTOK);
}

extern "C" void kernel_launch(void* const* d_in, const int* in_sizes, int n_in,
                              void* d_out, int out_size, void* d_ws, size_t ws_size,
                              hipStream_t stream) {
  (void)in_sizes; (void)n_in; (void)out_size; (void)ws_size;
  const float* x      = (const float*)d_in[0];  // [B,S,D]
  const float* w_gate = (const float*)d_in[1];  // [D,E]
  const float* w_in   = (const float*)d_in[2];  // [E,2H,D]
  const float* w_out  = (const float*)d_in[3];  // [E,D,H]
  const float* bias   = (const float*)d_in[4];  // [D]
  float* out = (float*)d_out;                   // y [N,D] ++ loss [1]

  char* base = (char*)d_ws;
  int*   counts   = (int*)(base + 0);
  float* p_acc    = (float*)(base + 32);
  float* freq_acc = (float*)(base + 64);
  float* lse2     = (float*)(base + 96);
  float* wsum_in  = (float*)(base + 128);
  float* wsum_out = (float*)(base + 160);
  size_t off = 256;
  signed char* xq   = (signed char*)(base + off); off += (size_t)NTOK * D_;
  float* step_x     = (float*)(base + off);       off += (size_t)NTOK * 4;
  int*   list_tok   = (int*)(base + off);         off += (size_t)E_ * NTOK * 4;
  float* list_gate  = (float*)(base + off);       off += (size_t)E_ * NTOK * 4;
  signed char* winq = (signed char*)(base + off); off += (size_t)E_ * H2_ * D_;
  signed char* woutq= (signed char*)(base + off); off += (size_t)E_ * D_ * H_;
  float* hbuf       = (float*)(base + off);       off += (size_t)NTOK * H_ * 4;
  signed char* hq   = (signed char*)(base + off); off += (size_t)NTOK * H_;
  float* step_h     = (float*)(base + off);       off += (size_t)NTOK * 4;

  zero_accum_kernel<<<1, 64, 0, stream>>>((float*)base);

  router_kernel<<<NTOK / 8, 256, 0, stream>>>(x, w_gate, xq, step_x, counts,
                                              list_tok, list_gate, p_acc,
                                              freq_acc, lse2);

  wabs_kernel<<<dim3(64, E_), 256, 0, stream>>>(w_in, wsum_in, H2_ * D_);
  wabs_kernel<<<dim3(64, E_), 256, 0, stream>>>(w_out, wsum_out, D_ * H_);
  wquant_kernel<<<dim3(256, E_), 256, 0, stream>>>(w_in, winq, wsum_in, H2_ * D_);
  wquant_kernel<<<dim3(256, E_), 256, 0, stream>>>(w_out, woutq, wsum_out, D_ * H_);

  init_y_kernel<<<(NTOK * D_) / 256, 256, 0, stream>>>(out, bias);

  for (int e = 0; e < E_; ++e) {
    // worst case: all 8192 tokens to one expert
    gemm1_glu_kernel<<<64 * (H_ / 64), 256, 0, stream>>>(
        xq, step_x, winq, wsum_in, counts, list_tok, hbuf, e);
    hquant_kernel<<<NTOK / 8, 256, 0, stream>>>(hbuf, hq, step_h, counts, e);
    gemm2_kernel<<<32 * (D_ / 64), 256, 0, stream>>>(
        hq, step_h, woutq, wsum_out, counts, list_tok, list_gate, out, e);
  }

  loss_kernel<<<1, 1, 0, stream>>>(p_acc, freq_acc, lse2, out);
}
